// BirdCLEFEncoder_137438953876
// MI455X (gfx1250) — matrix operand order repaired and checked by reading, not compiled, not run on hardware
//
#include <hip/hip_runtime.h>

typedef __bf16 bf16;
typedef __attribute__((ext_vector_type(16))) __bf16 v16bf;
typedef __attribute__((ext_vector_type(8)))  __bf16 v8bf;
typedef __attribute__((ext_vector_type(8)))  float  v8f;
typedef int v4i_ __attribute__((vector_size(16)));

#define WMMA_BF16(a,b,c) __builtin_amdgcn_wmma_f32_16x16x32_bf16(false,(a),false,(b),(short)0,(c),false,false)

#define BATCH 16
#define TT    1024
#define DDIM  256
#define MTOK  (BATCH*TT)      // 16384 tokens
#define EPSF  1e-5f
#define ASTR  40              // LDS stride for A tile rows (bf16), bank-conflict pad, 16B-aligned
#define CBSTR 136             // conv B tile row stride (bf16), 272B, 16B-aligned
#define GBSTR 264             // gemm B tile row stride (bf16), 528B, 16B-aligned

#if defined(__has_builtin)
#if __has_builtin(__builtin_amdgcn_global_load_async_to_lds_b128) && \
    __has_builtin(__builtin_amdgcn_s_wait_asynccnt)
#define USE_ASYNC 1
#endif
#endif

__device__ __forceinline__ float gelu_erf(float x) {
  return 0.5f * x * (1.0f + erff(x * 0.70710678118654752f));
}

union FragU { v16bf v; v8bf h[2]; };

// A fragment 16x32 bf16: lanes 0-15 -> M=lane, K={0..7,16..23}; lanes 16-31 -> K={8..15,24..31}
__device__ __forceinline__ v16bf load_a_frag(const bf16* lds, int m0) {
  int lane = threadIdx.x & 31;
  int half = lane >> 4, r = lane & 15;
  const bf16* p = lds + (m0 + r) * ASTR + half * 8;
  FragU u;
  u.h[0] = *(const v8bf*)(p);
  u.h[1] = *(const v8bf*)(p + 16);
  return u.v;
}

// B fragment 32x16 bf16: lanes 0-15 -> N=lane, K={0..7,16..23}; lanes 16-31 -> K={8..15,24..31}
__device__ __forceinline__ v16bf load_b_frag(const bf16* lds, int stride, int n0) {
  int lane = threadIdx.x & 31;
  int half = lane >> 4, c = lane & 15;
  const bf16* p = lds + half * 8 * stride + n0 + c;
  v16bf v;
  for (int s = 0; s < 8; ++s) {
    v[s] = p[s * stride];
    v[s + 8] = p[(s + 16) * stride];
  }
  return v;
}

#if defined(USE_ASYNC)
// GLOBAL_LOAD_ASYNC_TO_LDS_B128: (AS1 v4i* gaddr, AS3 v4i* ldsaddr, imm offset, imm cpol)
__device__ __forceinline__ void async_b128(const void* gsrc, void* ldst) {
  __builtin_amdgcn_global_load_async_to_lds_b128(
      (__attribute__((address_space(1))) v4i_*)(unsigned long long)gsrc,
      (__attribute__((address_space(3))) v4i_*)(unsigned)(unsigned long long)ldst,
      0, 0);
}
#endif

// ---------------------------------------------------------------------------
// GEMM: C[M,N] = A[M,K](bf16) @ B[K,N](bf16) + bias, opt GELU, opt residual.
// Block tile 64x256, 8 waves (2x4), wave tile 32x64 (8 WMMA accumulators).
// Double-buffered LDS stage fed by GLOBAL_LOAD_ASYNC_TO_LDS_B128 (ASYNCcnt).
// ---------------------------------------------------------------------------
__global__ __launch_bounds__(256) void k_gemm(
    const bf16* __restrict__ A, const bf16* __restrict__ Bkn,
    int K, int N,
    const float* __restrict__ bias, int act,
    float* __restrict__ outF, bf16* __restrict__ outBF,
    const float* __restrict__ resid)
{
  __shared__ bf16 sA[2][64 * ASTR];
  __shared__ bf16 sB[2][32 * GBSTR];
  const int m0 = blockIdx.y * 64;
  const int n0 = blockIdx.x * 256;
  const int tid = threadIdx.x;
  const int wave = tid >> 5;
  const int wm = (wave >> 2) * 32;       // 0 or 32
  const int wn = (wave & 3) * 64;        // 0,64,128,192

  // per-thread staging chunks: A 1x v8bf, B 4x v8bf
  const int arow = tid >> 2, acol = (tid & 3) * 8;

  v8f acc[2][4];
  for (int ti = 0; ti < 2; ++ti)
    for (int tj = 0; tj < 4; ++tj)
      acc[ti][tj] = (v8f){0.f,0.f,0.f,0.f,0.f,0.f,0.f,0.f};

  auto issue_tile = [&](int buf, int kc) {
#if defined(USE_ASYNC)
    async_b128(A + (size_t)(m0 + arow) * K + kc + acol, &sA[buf][arow * ASTR + acol]);
    for (int u = 0; u < 4; ++u) {
      int ch = tid * 4 + u;
      int kk = ch >> 5, nn = (ch & 31) * 8;
      async_b128(Bkn + (size_t)(kc + kk) * N + n0 + nn, &sB[buf][kk * GBSTR + nn]);
    }
#else
    *(v8bf*)(&sA[buf][arow * ASTR + acol]) =
        *(const v8bf*)(A + (size_t)(m0 + arow) * K + kc + acol);
    for (int u = 0; u < 4; ++u) {
      int ch = tid * 4 + u;
      int kk = ch >> 5, nn = (ch & 31) * 8;
      *(v8bf*)(&sB[buf][kk * GBSTR + nn]) =
          *(const v8bf*)(Bkn + (size_t)(kc + kk) * N + n0 + nn);
    }
#endif
  };

  const int nk = K >> 5;
  issue_tile(0, 0);
  for (int k = 0; k < nk; ++k) {
    const int cur = k & 1;
    const bool has_next = (k + 1) < nk;
    if (has_next) issue_tile(1 - cur, (k + 1) << 5);
#if defined(USE_ASYNC)
    if (has_next) __builtin_amdgcn_s_wait_asynccnt(5);  // current tile done (in-order), next in flight
    else          __builtin_amdgcn_s_wait_asynccnt(0);
#endif
    __syncthreads();
    v16bf a0 = load_a_frag(sA[cur], wm);
    v16bf a1 = load_a_frag(sA[cur], wm + 16);
    v16bf b0 = load_b_frag(sB[cur], GBSTR, wn);
    v16bf b1 = load_b_frag(sB[cur], GBSTR, wn + 16);
    v16bf b2 = load_b_frag(sB[cur], GBSTR, wn + 32);
    v16bf b3 = load_b_frag(sB[cur], GBSTR, wn + 48);
    acc[0][0] = WMMA_BF16(a0, b0, acc[0][0]);
    acc[0][1] = WMMA_BF16(a0, b1, acc[0][1]);
    acc[0][2] = WMMA_BF16(a0, b2, acc[0][2]);
    acc[0][3] = WMMA_BF16(a0, b3, acc[0][3]);
    acc[1][0] = WMMA_BF16(a1, b0, acc[1][0]);
    acc[1][1] = WMMA_BF16(a1, b1, acc[1][1]);
    acc[1][2] = WMMA_BF16(a1, b2, acc[1][2]);
    acc[1][3] = WMMA_BF16(a1, b3, acc[1][3]);
    __syncthreads();   // all waves done with buf[cur] before it is re-filled
  }

  // epilogue: C/D layout -> lane<16: N=lane, M=i; lane>=16: N=lane-16, M=8+i
  const int lane = tid & 31, half = lane >> 4, r = lane & 15;
  for (int ti = 0; ti < 2; ++ti) {
    for (int tj = 0; tj < 4; ++tj) {
      int col = n0 + wn + tj * 16 + r;
      float bv = bias ? bias[col] : 0.f;
      for (int i = 0; i < 8; ++i) {
        int row = m0 + wm + ti * 16 + half * 8 + i;
        float v = acc[ti][tj][i] + bv;
        if (resid) v += resid[(size_t)row * N + col];
        if (act == 1) v = gelu_erf(v);
        if (outF)  outF[(size_t)row * N + col] = v;
        if (outBF) outBF[(size_t)row * N + col] = (bf16)v;
      }
    }
  }
}

// ---------------------------------------------------------------------------
// Implicit-GEMM conv (stride (2,1)) + BN + exact GELU, WMMA core.
// M = Cout, N = 128 t-positions per block, K = Cin*k*k (zero-padded to %32).
// ---------------------------------------------------------------------------
__global__ __launch_bounds__(256) void k_conv_wmma(
    const bf16* __restrict__ X, const bf16* __restrict__ Wp,
    bf16* __restrict__ Y,
    const float* __restrict__ cb, const float* __restrict__ bg,
    const float* __restrict__ bbp, const float* __restrict__ bm,
    const float* __restrict__ bvv,
    int Cin, int Cout, int ksz, int pad, int Fin, int Fout,
    int Kreal, int Kpad, int mtiles)
{
  __shared__ bf16 sA[64 * ASTR];
  __shared__ bf16 sB[32 * CBSTR];
  const int t0  = blockIdx.x * 128;
  const int f   = blockIdx.y;
  const int co0 = (blockIdx.z % mtiles) * 64;
  const int nb  = blockIdx.z / mtiles;
  const int tid = threadIdx.x;
  const int wave = tid >> 5;
  const int wm = (wave >> 2) * 32;
  const int wn = (wave & 3) * 32;
  const int kk2 = ksz * ksz;

  v8f c00 = {0.f,0.f,0.f,0.f,0.f,0.f,0.f,0.f};
  v8f c01 = c00, c10 = c00, c11 = c00;

  for (int kc = 0; kc < Kpad; kc += 32) {
    { // weights tile (pre-packed, zero padded)
      int row = tid >> 2, col = (tid & 3) * 8;
      v8bf v = *(const v8bf*)(Wp + (size_t)(co0 + row) * Kpad + kc + col);
      *(v8bf*)(sA + row * ASTR + col) = v;
      if (kc + 32 < Kpad)   // hint next weight chunk into cache (global_prefetch_b8)
        __builtin_prefetch(Wp + (size_t)(co0 + row) * Kpad + kc + 32 + col, 0, 1);
    }
    // im2col gather for B tile 32(K) x 128(t)
    for (int e = tid; e < 32 * 128; e += 256) {
      int kkl = e >> 7, nn = e & 127;
      int kk = kc + kkl;
      float val = 0.f;
      if (kk < Kreal) {
        int ci  = kk / kk2;
        int rsm = kk - ci * kk2;
        int kh  = rsm / ksz;
        int kw  = rsm - kh * ksz;
        int fin = 2 * f + kh - pad;
        int tin = t0 + nn + kw - pad;
        if (fin >= 0 && fin < Fin && tin >= 0 && tin < TT)
          val = (float)X[(((size_t)nb * Cin + ci) * Fin + fin) * TT + tin];
      }
      sB[kkl * CBSTR + nn] = (bf16)val;
    }
    __syncthreads();
    v16bf a0 = load_a_frag(sA, wm);
    v16bf a1 = load_a_frag(sA, wm + 16);
    v16bf b0 = load_b_frag(sB, CBSTR, wn);
    v16bf b1 = load_b_frag(sB, CBSTR, wn + 16);
    c00 = WMMA_BF16(a0, b0, c00);
    c01 = WMMA_BF16(a0, b1, c01);
    c10 = WMMA_BF16(a1, b0, c10);
    c11 = WMMA_BF16(a1, b1, c11);
    __syncthreads();
  }

  const int lane = tid & 31, half = lane >> 4, r = lane & 15;
  v8f acc[2][2] = {{c00, c01}, {c10, c11}};
  for (int ti = 0; ti < 2; ++ti) {
    for (int tj = 0; tj < 2; ++tj) {
      int t = t0 + wn + tj * 16 + r;
      for (int i = 0; i < 8; ++i) {
        int co = co0 + wm + ti * 16 + half * 8 + i;
        if (co < Cout) {
          float scale = bg[co] * rsqrtf(bvv[co] + EPSF);
          float shift = (cb[co] - bm[co]) * scale + bbp[co];
          float v = gelu_erf(acc[ti][tj][i] * scale + shift);
          Y[(((size_t)nb * Cout + co) * Fout + f) * TT + t] = (bf16)v;
        }
      }
    }
  }
}

// ---------------------------------------------------------------------------
// Small utility / VALU kernels
// ---------------------------------------------------------------------------
__global__ void k_convert_bf16(const float* __restrict__ s, bf16* __restrict__ d, int n) {
  int i = blockIdx.x * 256 + threadIdx.x;
  if (i < n) d[i] = (bf16)s[i];
}

__global__ void k_pack_conv_w(const float* __restrict__ w, bf16* __restrict__ dst,
                              int Cout, int Kreal, int Mpad, int Kpad) {
  int i = blockIdx.x * 256 + threadIdx.x;
  if (i >= Mpad * Kpad) return;
  int co = i / Kpad, kk = i - co * Kpad;
  float v = (co < Cout && kk < Kreal) ? w[(size_t)co * Kreal + kk] : 0.f;
  dst[i] = (bf16)v;
}

// src [R][C] f32 -> dst [C][R] bf16
__global__ void k_pack_transpose(const float* __restrict__ src, bf16* __restrict__ dst, int R, int C) {
  int i = blockIdx.x * 256 + threadIdx.x;
  if (i >= R * C) return;
  int c = i % C, rr = i / C;
  dst[(size_t)c * R + rr] = (bf16)src[i];
}

// conv3 out [B][128][8][1024] bf16 -> token-major [b*1024+t][c*8+f]
__global__ void k_token_pack(const bf16* __restrict__ c3, bf16* __restrict__ Atok) {
  size_t i = (size_t)blockIdx.x * 256 + threadIdx.x;
  if (i >= (size_t)MTOK * 1024) return;
  int k = (int)(i & 1023);
  size_t tok = i >> 10;
  int b = (int)(tok >> 10), t = (int)(tok & 1023);
  int c = k >> 3, fidx = k & 7;
  Atok[i] = c3[(((size_t)b * 128 + c) * 8 + fidx) * 1024 + t];
}

__global__ void k_posenc_add(float* __restrict__ x) {
  int i = blockIdx.x * 256 + threadIdx.x;
  if (i >= MTOK * DDIM) return;
  int d = i & 255;
  int t = (i >> 8) & 1023;
  int p = d >> 1;
  float ang = (float)t * __expf((float)(2 * p) * (-9.2103403719761836f / 256.0f));
  x[i] += (d & 1) ? __cosf(ang) : __sinf(ang);
}

__global__ __launch_bounds__(256) void k_layernorm(const float* __restrict__ x,
    const float* __restrict__ g, const float* __restrict__ b, bf16* __restrict__ out) {
  int wave = threadIdx.x >> 5, lane = threadIdx.x & 31;
  int row = blockIdx.x * 8 + wave;
  const float* xr = x + (size_t)row * DDIM;
  float v[8]; float s = 0.f, sq = 0.f;
  for (int j = 0; j < 8; ++j) { v[j] = xr[lane + 32 * j]; s += v[j]; sq += v[j] * v[j]; }
  for (int off = 16; off; off >>= 1) { s += __shfl_xor(s, off); sq += __shfl_xor(sq, off); }
  float mu = s * (1.f / 256.f);
  float var = sq * (1.f / 256.f) - mu * mu;
  float rstd = rsqrtf(var + EPSF);
  for (int j = 0; j < 8; ++j) {
    int d = lane + 32 * j;
    out[(size_t)row * DDIM + d] = (bf16)((v[j] - mu) * rstd * g[d] + b[d]);
  }
}

// Block-0 attention: window +-8 -> one wave per (b,h,i); lanes 0..16 = keys, then lanes = dims.
__global__ __launch_bounds__(256) void k_attn_local(const float* __restrict__ qkv,
                                                    bf16* __restrict__ attn) {
  int gw = blockIdx.x * 8 + (threadIdx.x >> 5);
  int lane = threadIdx.x & 31;
  int i = gw & 1023;
  int h = (gw >> 10) & 7;
  int b = gw >> 13;
  const float* base = qkv + (size_t)b * 1024 * 768;
  const float* q = base + (size_t)i * 768 + h * 32;
  int j = lane, jj = i - 8 + j;
  bool valid = (j < 17) && (jj >= 0) && (jj < 1024);
  float s = -3.0e38f;
  if (valid) {
    const float* kp = base + (size_t)jj * 768 + 256 + h * 32;
    float a = 0.f;
    for (int d = 0; d < 32; ++d) a += q[d] * kp[d];
    s = a * 0.17677669529663687f;
  }
  float m = s;
  for (int off = 16; off; off >>= 1) m = fmaxf(m, __shfl_xor(m, off));
  float p = valid ? __expf(s - m) : 0.f;
  float l = p;
  for (int off = 16; off; off >>= 1) l += __shfl_xor(l, off);
  float o = 0.f;  // lane = dim now
  for (int j2 = 0; j2 < 17; ++j2) {
    int jj2 = i - 8 + j2;
    float pj = __shfl(p, j2);
    if (jj2 >= 0 && jj2 < 1024)
      o += pj * base[(size_t)jj2 * 768 + 512 + h * 32 + lane];
  }
  attn[(size_t)(b * 1024 + i) * 256 + h * 32 + lane] = (bf16)(o / l);
}

// Block-1 attention, non-global rows: softmax over {self} -> o = v_i.
__global__ void k_attn_diag(const float* __restrict__ qkv, bf16* __restrict__ attn) {
  size_t idx = (size_t)blockIdx.x * 256 + threadIdx.x;
  if (idx >= (size_t)MTOK * 256) return;
  int d = (int)(idx & 255);
  size_t tok = idx >> 8;
  int i = (int)(tok & 1023);
  if (i % 100 == 0) return;  // global rows handled below
  attn[idx] = (bf16)qkv[tok * 768 + 512 + d];
}

// Block-1 attention, 11 global rows per (b,h): full softmax over 1024 keys; one wave each.
__global__ __launch_bounds__(32) void k_attn_grow(const float* __restrict__ qkv,
                                                  bf16* __restrict__ attn) {
  __shared__ float sc[1024];
  int gi = blockIdx.x % 11;
  int h  = (blockIdx.x / 11) % 8;
  int b  = blockIdx.x / 88;
  int i = gi * 100;
  int lane = threadIdx.x;
  const float* base = qkv + (size_t)b * 1024 * 768;
  const float* q = base + (size_t)i * 768 + h * 32;
  float m = -3.0e38f;
  for (int j = lane; j < 1024; j += 32) {
    const float* kp = base + (size_t)j * 768 + 256 + h * 32;
    float a = 0.f;
    for (int d = 0; d < 32; ++d) a += q[d] * kp[d];
    a *= 0.17677669529663687f;
    sc[j] = a;
    m = fmaxf(m, a);
  }
  for (int off = 16; off; off >>= 1) m = fmaxf(m, __shfl_xor(m, off));
  __syncthreads();
  float l = 0.f;
  for (int j = lane; j < 1024; j += 32) {
    float e = __expf(sc[j] - m);
    sc[j] = e;
    l += e;
  }
  for (int off = 16; off; off >>= 1) l += __shfl_xor(l, off);
  __syncthreads();
  float o = 0.f;  // lane = dim
  for (int j = 0; j < 1024; ++j)
    o += sc[j] * base[(size_t)j * 768 + 512 + h * 32 + lane];
  attn[(size_t)(b * 1024 + i) * 256 + h * 32 + lane] = (bf16)(o / l);
}

__global__ void k_copy_f32(const float* __restrict__ s, float* __restrict__ d, int n) {
  int i = blockIdx.x * 256 + threadIdx.x;
  if (i < n) d[i] = s[i];
}

// ---------------------------------------------------------------------------
extern "C" void kernel_launch(void* const* d_in, const int* in_sizes, int n_in,
                              void* d_out, int out_size, void* d_ws, size_t ws_size,
                              hipStream_t stream) {
  (void)in_sizes; (void)n_in; (void)out_size; (void)ws_size;
  auto P = [&](int i) { return (const float*)d_in[i]; };
  const float* x      = P(0);
  const float* proj_w = P(25);
  const float* proj_b = P(26);

  char* ws = (char*)d_ws;
  size_t off = 0;
  auto alloc = [&](size_t bytes) { size_t o = off; off += (bytes + 255) & ~(size_t)255; return o; };
  size_t o_xbf  = alloc((size_t)2097152 * 2);
  size_t o_wp0  = alloc((size_t)64 * 32 * 2);
  size_t o_wp1  = alloc((size_t)64 * 800 * 2);
  size_t o_wp2  = alloc((size_t)128 * 3136 * 2);
  size_t o_wp3  = alloc((size_t)128 * 6272 * 2);
  size_t o_btp  = alloc((size_t)1024 * 256 * 2);
  size_t o_btin = alloc((size_t)256 * 768 * 2);
  size_t o_btout= alloc((size_t)256 * 256 * 2);
  size_t o_btw1 = alloc((size_t)256 * 1024 * 2);
  size_t o_btw2 = alloc((size_t)1024 * 256 * 2);
  size_t o_bigA = alloc((size_t)67108864);   // conv0/conv2 acts; later qkv f32 (50.3MB)
  size_t o_bigB = alloc((size_t)67108864);   // conv1/conv3 acts; later xln/attn/hid
  size_t o_atok = alloc((size_t)MTOK * 1024 * 2);
  size_t o_xres = alloc((size_t)MTOK * 256 * 4);

  bf16* xbf   = (bf16*)(ws + o_xbf);
  bf16* wp[4] = {(bf16*)(ws + o_wp0), (bf16*)(ws + o_wp1), (bf16*)(ws + o_wp2), (bf16*)(ws + o_wp3)};
  bf16* btp   = (bf16*)(ws + o_btp);
  bf16* btin  = (bf16*)(ws + o_btin);
  bf16* btout = (bf16*)(ws + o_btout);
  bf16* btw1  = (bf16*)(ws + o_btw1);
  bf16* btw2  = (bf16*)(ws + o_btw2);
  bf16* act0  = (bf16*)(ws + o_bigA);
  bf16* act1  = (bf16*)(ws + o_bigB);
  bf16* act2  = (bf16*)(ws + o_bigA);
  bf16* act3  = (bf16*)(ws + o_bigB);
  float* qkv  = (float*)(ws + o_bigA);
  bf16* xln   = (bf16*)(ws + o_bigB);
  bf16* attnb = (bf16*)(ws + o_bigB + ((size_t)16 << 20));
  bf16* hid   = (bf16*)(ws + o_bigB + ((size_t)32 << 20));
  bf16* Atok  = (bf16*)(ws + o_atok);
  float* xres = (float*)(ws + o_xres);
  float* out  = (float*)d_out;

  // input -> bf16
  k_convert_bf16<<<(2097152 + 255) / 256, 256, 0, stream>>>(x, xbf, 2097152);

  // conv weights pack (zero-padded to Mpad x Kpad)
  const int CoutA[4] = {32, 64, 128, 128};
  const int CinA[4]  = {1, 32, 64, 128};
  const int KszA[4]  = {3, 5, 7, 7};
  const int PadA[4]  = {1, 2, 3, 3};
  const int KrealA[4]= {9, 800, 3136, 6272};
  const int KpadA[4] = {32, 800, 3136, 6272};
  const int MpadA[4] = {64, 64, 128, 128};
  const int FinA[4]  = {128, 64, 32, 16};
  const int FoutA[4] = {64, 32, 16, 8};
  for (int li = 0; li < 4; ++li) {
    int tot = MpadA[li] * KpadA[li];
    k_pack_conv_w<<<(tot + 255) / 256, 256, 0, stream>>>(
        P(1 + 6 * li), wp[li], CoutA[li], KrealA[li], MpadA[li], KpadA[li]);
  }

  // conv stack
  const bf16* cin[4] = {xbf, act0, act1, act2};
  bf16* cout[4] = {act0, act1, act2, act3};
  for (int li = 0; li < 4; ++li) {
    int mtiles = MpadA[li] / 64;
    dim3 g(TT / 128, FoutA[li], BATCH * mtiles);
    k_conv_wmma<<<g, 256, 0, stream>>>(
        cin[li], wp[li], cout[li],
        P(2 + 6 * li), P(3 + 6 * li), P(4 + 6 * li), P(5 + 6 * li), P(6 + 6 * li),
        CinA[li], CoutA[li], KszA[li], PadA[li], FinA[li], FoutA[li],
        KrealA[li], KpadA[li], mtiles);
  }

  // tokens + projection + positional encoding
  k_token_pack<<<(int)(((size_t)MTOK * 1024 + 255) / 256), 256, 0, stream>>>(act3, Atok);
  k_convert_bf16<<<(262144 + 255) / 256, 256, 0, stream>>>(proj_w, btp, 262144);  // already [K][N]
  k_gemm<<<dim3(256 / 256, MTOK / 64), 256, 0, stream>>>(
      Atok, btp, 1024, 256, proj_b, 0, xres, nullptr, nullptr);
  k_posenc_add<<<(MTOK * DDIM) / 256, 256, 0, stream>>>(xres);

  // transformer blocks
  for (int bi = 0; bi < 2; ++bi) {
    int base = 27 + 12 * bi;
    const float* in_w  = P(base + 0);  const float* in_b  = P(base + 1);
    const float* out_w = P(base + 2);  const float* out_b = P(base + 3);
    const float* ln1g  = P(base + 4);  const float* ln1b  = P(base + 5);
    const float* ln2g  = P(base + 6);  const float* ln2b  = P(base + 7);
    const float* w1    = P(base + 8);  const float* b1    = P(base + 9);
    const float* w2    = P(base + 10); const float* b2    = P(base + 11);

    k_pack_transpose<<<(768 * 256 + 255) / 256, 256, 0, stream>>>(in_w, btin, 768, 256);
    k_pack_transpose<<<(256 * 256 + 255) / 256, 256, 0, stream>>>(out_w, btout, 256, 256);
    k_convert_bf16<<<(262144 + 255) / 256, 256, 0, stream>>>(w1, btw1, 262144);  // [256][1024]
    k_convert_bf16<<<(262144 + 255) / 256, 256, 0, stream>>>(w2, btw2, 262144);  // [1024][256]

    k_layernorm<<<MTOK / 8, 256, 0, stream>>>(xres, ln1g, ln1b, xln);
    k_gemm<<<dim3(768 / 256, MTOK / 64), 256, 0, stream>>>(
        xln, btin, 256, 768, in_b, 0, qkv, nullptr, nullptr);

    if (bi == 0) {
      k_attn_local<<<(BATCH * 8 * TT) / 8, 256, 0, stream>>>(qkv, attnb);
    } else {
      k_attn_diag<<<(int)(((size_t)MTOK * 256) / 256), 256, 0, stream>>>(qkv, attnb);
      k_attn_grow<<<BATCH * 8 * 11, 32, 0, stream>>>(qkv, attnb);
    }

    k_gemm<<<dim3(256 / 256, MTOK / 64), 256, 0, stream>>>(
        attnb, btout, 256, 256, out_b, 0, xres, nullptr, xres);

    k_layernorm<<<MTOK / 8, 256, 0, stream>>>(xres, ln2g, ln2b, xln);
    k_gemm<<<dim3(1024 / 256, MTOK / 64), 256, 0, stream>>>(
        xln, btw1, 256, 1024, b1, 1, nullptr, hid, nullptr);
    k_gemm<<<dim3(256 / 256, MTOK / 64), 256, 0, stream>>>(
        hid, btw2, 1024, 256, b2, 0, xres, nullptr, xres);

    // outputs: [final | inter0 | inter1]; inter1 == final
    int nelem = MTOK * DDIM;
    if (bi == 0) {
      k_copy_f32<<<nelem / 256, 256, 0, stream>>>(xres, out + nelem, nelem);
    } else {
      k_copy_f32<<<nelem / 256, 256, 0, stream>>>(xres, out, nelem);
      k_copy_f32<<<nelem / 256, 256, 0, stream>>>(xres, out + 2 * (size_t)nelem, nelem);
    }
  }
}
